// HeteroGNN_17179869368
// MI455X (gfx1250) — matrix operand order repaired and checked by reading, not compiled, not run on hardware
//
#include <hip/hip_runtime.h>
#include <stdint.h>

#define HID 128
#define HH  256   // H * HID
#define SBERT 384

typedef __attribute__((ext_vector_type(16))) __bf16 v16bf;
typedef __attribute__((ext_vector_type(8)))  float  v8f;

union Frag { v16bf v; unsigned u[8]; };

__device__ __forceinline__ unsigned pack_bf2(float a, float b) {
  union { __bf16 h[2]; unsigned u; } c;
  c.h[0] = (__bf16)a;   // native v_cvt_pk_bf16_f32
  c.h[1] = (__bf16)b;
  return c.u;
}

// Order-preserving float<->uint keys for atomic max over floats.
__device__ __forceinline__ unsigned okey(float f) {
  unsigned u = __float_as_uint(f);
  return (u & 0x80000000u) ? ~u : (u | 0x80000000u);
}
__device__ __forceinline__ float dkey(unsigned k) {
  unsigned b = (k & 0x80000000u) ? (k ^ 0x80000000u) : ~k;
  return __uint_as_float(b);
}

// ---------------------------------------------------------------------------
// Repack f32 weight W[K,N] (row-major) into bf16 B-fragment layout:
// fragment (kt,nt) occupies 256 dwords; lane holds 8 contiguous dwords;
// dword j of lane l = bf16{ W[kt*32+((l>>4)<<4)+2j][nt*16+(l&15)], W[..+1][..] }.
// ---------------------------------------------------------------------------
__global__ void repack_w(const float* __restrict__ W, unsigned* __restrict__ out,
                         int K, int N) {
  int idx = blockIdx.x * blockDim.x + threadIdx.x;
  int total = (K >> 5) * (N >> 4) * 256;
  if (idx >= total) return;
  int j    = idx & 7;
  int lane = (idx >> 3) & 31;
  int frag = idx >> 8;
  int nt = frag % (N >> 4);
  int kt = frag / (N >> 4);
  int n = nt * 16 + (lane & 15);
  int k = kt * 32 + ((lane >> 4) << 4) + 2 * j;
  out[idx] = pack_bf2(W[(size_t)k * N + n], W[(size_t)(k + 1) * N + n]);
}

// ---------------------------------------------------------------------------
// WMMA GEMM: out[M,N] = epilogue(A[M,K] @ W[K,N] (+bias)); W pre-packed bf16,
// staged through LDS in 32KB chunks (shared by all 4 waves of the block).
// EPI: 0 = (optional bias), 1 = bias+ReLU, 2 = bias+LayerNorm+ReLU (N==128)
// Block = 128 threads = 4 waves; each wave owns a 16-row block and full N.
// ---------------------------------------------------------------------------
template<int K, int N, int EPI>
__launch_bounds__(128)
__global__ void wmma_gemm(const float* __restrict__ A, const unsigned* __restrict__ Wp,
                          const float* __restrict__ bias, const float* __restrict__ gamma,
                          const float* __restrict__ beta, float* __restrict__ out, int M) {
  constexpr int NT = N / 16;
  constexpr int KT = K / 32;
  constexpr int KC = (NT == 16) ? 2 : 4;     // K-tiles per LDS chunk
  constexpr int CH = KT / KC;                // chunks
  constexpr int CHUNK_DW = KC * NT * 256;    // 8192 dwords = 32KB per chunk
  __shared__ unsigned sh[CHUNK_DW];

  const int lane = threadIdx.x & 31;
  const int wave = threadIdx.x >> 5;
  const int row0 = (blockIdx.x * 4 + wave) * 16;
  const bool active = (row0 < M);
  const int row0c = active ? row0 : 0;       // clamp so inactive waves stay in-bounds

  v8f acc[NT];
#pragma unroll
  for (int t = 0; t < NT; ++t) {
    v8f z = {0.f, 0.f, 0.f, 0.f, 0.f, 0.f, 0.f, 0.f};
    acc[t] = z;
  }

  const int mrow = row0c + (lane & 15);
  const float* arow = A + (size_t)mrow * K + ((lane >> 4) << 4);
  const unsigned* shlane = sh + lane * 8;

  for (int ch = 0; ch < CH; ++ch) {
    __syncthreads();
    {  // cooperative global -> LDS stage of this chunk's fragments
      const uint4* g  = (const uint4*)(Wp + (size_t)ch * CHUNK_DW);
      uint4*       s4 = (uint4*)sh;
#pragma unroll
      for (int i = 0; i < CHUNK_DW / 4 / 128; ++i)
        s4[i * 128 + threadIdx.x] = g[i * 128 + threadIdx.x];
    }
    __syncthreads();

#pragma unroll
    for (int kc = 0; kc < KC; ++kc) {
      const int kt = ch * KC + kc;
      // A fragment (lanes 0-15: K 32kt+0..15; lanes 16-31: +16)
      const float4 a0 = *(const float4*)(arow + kt * 32 + 0);
      const float4 a1 = *(const float4*)(arow + kt * 32 + 4);
      const float4 a2 = *(const float4*)(arow + kt * 32 + 8);
      const float4 a3 = *(const float4*)(arow + kt * 32 + 12);
      Frag fa;
      fa.u[0] = pack_bf2(a0.x, a0.y);  fa.u[1] = pack_bf2(a0.z, a0.w);
      fa.u[2] = pack_bf2(a1.x, a1.y);  fa.u[3] = pack_bf2(a1.z, a1.w);
      fa.u[4] = pack_bf2(a2.x, a2.y);  fa.u[5] = pack_bf2(a2.z, a2.w);
      fa.u[6] = pack_bf2(a3.x, a3.y);  fa.u[7] = pack_bf2(a3.z, a3.w);

      const unsigned* wrow = shlane + kc * NT * 256;
#pragma unroll
      for (int t = 0; t < NT; ++t) {
        Frag fb;
        *(uint4*)&fb.u[0] = *(const uint4*)(wrow + t * 256);
        *(uint4*)&fb.u[4] = *(const uint4*)(wrow + t * 256 + 4);
        acc[t] = __builtin_amdgcn_wmma_f32_16x16x32_bf16(
            false, fa.v, false, fb.v, (short)0, acc[t], false, false);
      }
    }
  }

  if (!active) return;

  // Epilogue. C layout: VGPR r -> row row0+r (lanes 0-15) / row0+r+8 (lanes 16-31),
  // col = t*16 + (lane&15). Hoist per-column parameters out of the row loop.
  const int colb  = lane & 15;
  const int rbase = (lane >> 4) << 3;
  float bvals[NT];
  if (bias) {
#pragma unroll
    for (int t = 0; t < NT; ++t) bvals[t] = bias[t * 16 + colb];
  } else {
#pragma unroll
    for (int t = 0; t < NT; ++t) bvals[t] = 0.f;
  }
  float gv[NT], bev[NT];
  if (EPI == 2) {
#pragma unroll
    for (int t = 0; t < NT; ++t) { gv[t] = gamma[t * 16 + colb]; bev[t] = beta[t * 16 + colb]; }
  }

#pragma unroll
  for (int r = 0; r < 8; ++r) {
    const int row = row0 + rbase + r;
    float vals[NT];
#pragma unroll
    for (int t = 0; t < NT; ++t) vals[t] = acc[t][r] + bvals[t];
    if (EPI == 2) {
      float s1 = 0.f, s2 = 0.f;
#pragma unroll
      for (int t = 0; t < NT; ++t) { s1 += vals[t]; s2 += vals[t] * vals[t]; }
      s1 += __shfl_xor(s1, 1, 32);  s2 += __shfl_xor(s2, 1, 32);
      s1 += __shfl_xor(s1, 2, 32);  s2 += __shfl_xor(s2, 2, 32);
      s1 += __shfl_xor(s1, 4, 32);  s2 += __shfl_xor(s2, 4, 32);
      s1 += __shfl_xor(s1, 8, 32);  s2 += __shfl_xor(s2, 8, 32);
      const float mean = s1 / (float)N;
      const float var  = s2 / (float)N - mean * mean;
      const float rstd = rsqrtf(var + 1e-5f);
#pragma unroll
      for (int t = 0; t < NT; ++t) {
        float v = (vals[t] - mean) * rstd * gv[t] + bev[t];
        out[(size_t)row * N + t * 16 + colb] = fmaxf(v, 0.f);
      }
    } else {
#pragma unroll
      for (int t = 0; t < NT; ++t) {
        float v = vals[t];
        if (EPI == 1) v = fmaxf(v, 0.f);
        out[(size_t)row * N + t * 16 + colb] = v;
      }
    }
  }
}

// av[h*128+k] = sum_c Wd[k, h*128+c] * ad[h*128+c]   (Wd row-major [128,256])
__global__ void fold_vec(const float* __restrict__ Wd, const float* __restrict__ ad,
                         float* __restrict__ av) {
  int idx = blockIdx.x * blockDim.x + threadIdx.x;
  if (idx >= 256) return;
  int h = idx >> 7, k = idx & 127;
  const float* wp = Wd + (size_t)k * HH + h * HID;
  const float* ap = ad + h * HID;
  float s = 0.f;
#pragma unroll 4
  for (int c = 0; c < HID; ++c) s += wp[c] * ap[c];
  av[idx] = s;
}

// alpha[n,h] = sum_k in[n*istride + h*hoff + k] * vec[h*128+k]
__global__ void alpha_k(const float* __restrict__ in, int istride, int hoff,
                        const float* __restrict__ vec, float* __restrict__ alpha, int n2) {
  int idx = blockIdx.x * blockDim.x + threadIdx.x;
  if (idx >= n2) return;
  int node = idx >> 1, h = idx & 1;
  const float* ip = in + (size_t)node * istride + h * hoff;
  const float* vp = vec + h * HID;
  float s = 0.f;
#pragma unroll 4
  for (int k = 0; k < HID; ++k) s += ip[k] * vp[k];
  alpha[idx] = s;
}

__global__ void fill_u32(unsigned* __restrict__ p, unsigned v, int n) {
  int i = blockIdx.x * blockDim.x + threadIdx.x;
  if (i < n) p[i] = v;
}

// Pass 1: scatter-max of leaky_relu(alpha_s[src]+alpha_d[dst]) per (dst, head).
__global__ void edge_max(const int* __restrict__ src, const int* __restrict__ dst,
                         const float* __restrict__ as_, const float* __restrict__ ad_,
                         unsigned* __restrict__ mkey, int E, int Etot) {
  int idx = blockIdx.x * blockDim.x + threadIdx.x;
  if (idx >= Etot * 2) return;
  int e = idx >> 1, h = idx & 1;
  int s = (e < E) ? src[e] : (e - E);
  int d = (e < E) ? dst[e] : (e - E);
  float x = as_[s * 2 + h] + ad_[d * 2 + h];
  x = (x > 0.f) ? x : 0.2f * x;
  atomicMax(&mkey[d * 2 + h], okey(x));
}

// Pass 2: p = exp(e - m[dst]); den[dst]+=p; acc[dst,:] += hs[src,:]*p. One wave/edge.
__launch_bounds__(256)
__global__ void edge_accum(const int* __restrict__ src, const int* __restrict__ dst,
                           const float* __restrict__ as_, const float* __restrict__ ad_,
                           const unsigned* __restrict__ mkey, float* __restrict__ den,
                           const float* __restrict__ hs, float* __restrict__ acc,
                           int E, int Etot) {
  int e = blockIdx.x * 8 + (threadIdx.x >> 5);
  if (e >= Etot) return;
  int lane = threadIdx.x & 31;
  int s = (e < E) ? src[e] : (e - E);
  int d = (e < E) ? dst[e] : (e - E);
  float x0 = as_[s * 2 + 0] + ad_[d * 2 + 0];
  x0 = (x0 > 0.f) ? x0 : 0.2f * x0;
  float p0 = __expf(x0 - dkey(mkey[d * 2 + 0]));
  float x1 = as_[s * 2 + 1] + ad_[d * 2 + 1];
  x1 = (x1 > 0.f) ? x1 : 0.2f * x1;
  float p1 = __expf(x1 - dkey(mkey[d * 2 + 1]));
  if (lane == 0) atomicAdd(&den[d * 2 + 0], p0);
  if (lane == 1) atomicAdd(&den[d * 2 + 1], p1);
  const float* hp = hs + (size_t)s * HH;
  float* op = acc + (size_t)d * HH;
#pragma unroll
  for (int i = 0; i < 8; ++i) {
    int c = i * 32 + lane;
    atomicAdd(&op[c], hp[c] * ((c < HID) ? p0 : p1));
  }
}

// out[n,c] = acc1/(den1+eps)+b1  (+ acc2/(den2+eps)+b2 for the summed skill relations)
__global__ void node_final(const float* __restrict__ acc1, const float* __restrict__ den1,
                           const float* __restrict__ b1,
                           const float* __restrict__ acc2, const float* __restrict__ den2,
                           const float* __restrict__ b2,
                           float* __restrict__ out, int n) {
  int idx = blockIdx.x * blockDim.x + threadIdx.x;
  if (idx >= n * HH) return;
  int node = idx >> 8, c = idx & 255, h = c >> 7;
  float v = acc1[idx] / (den1[node * 2 + h] + 1e-16f) + b1[c];
  if (acc2) v += acc2[idx] / (den2[node * 2 + h] + 1e-16f) + b2[c];
  out[idx] = v;
}

__global__ void qproj(const float* __restrict__ query, const float* __restrict__ Wq,
                      const float* __restrict__ bq, float* __restrict__ q) {
  int o = threadIdx.x;
  if (o >= 128) return;
  float s = bq[o];
  for (int k = 0; k < SBERT; ++k) s += query[k] * Wq[(size_t)k * 128 + o];
  q[o] = s;
}

__global__ void scores_k(const float* __restrict__ emb, const float* __restrict__ q,
                         float* __restrict__ sc, int nj) {
  int j = blockIdx.x * blockDim.x + threadIdx.x;
  if (j >= nj) return;
  const float* ep = emb + (size_t)j * 128;
  float s = 0.f;
#pragma unroll 4
  for (int k = 0; k < 128; ++k) s += ep[k] * q[k];
  sc[j] = s;
}

extern "C" void kernel_launch(void* const* d_in, const int* in_sizes, int n_in,
                              void* d_out, int out_size, void* d_ws, size_t ws_size,
                              hipStream_t stream) {
  (void)n_in; (void)out_size; (void)ws_size;
  const float* x_job   = (const float*)d_in[0];
  const float* x_skill = (const float*)d_in[1];
  const int* js_src = (const int*)d_in[2];
  const int* js_dst = (const int*)d_in[3];
  const int* sj_src = (const int*)d_in[4];
  const int* sj_dst = (const int*)d_in[5];
  const int* ss_src = (const int*)d_in[6];
  const int* ss_dst = (const int*)d_in[7];
  const float* query = (const float*)d_in[8];
  const float* W0j = (const float*)d_in[9];  const float* b0j  = (const float*)d_in[10];
  const float* g0j = (const float*)d_in[11]; const float* be0j = (const float*)d_in[12];
  const float* W0s = (const float*)d_in[13]; const float* b0s  = (const float*)d_in[14];
  const float* g0s = (const float*)d_in[15]; const float* be0s = (const float*)d_in[16];
  const float* gWs = (const float*)d_in[17];
  const float* gWd = (const float*)d_in[18];
  const float* gas = (const float*)d_in[19];
  const float* gad = (const float*)d_in[20];
  const float* gb  = (const float*)d_in[21];
  const float* iW  = (const float*)d_in[22];
  const float* ib  = (const float*)d_in[23];
  const float* Wjf = (const float*)d_in[24]; const float* bjf = (const float*)d_in[25];
  const float* Wq  = (const float*)d_in[26]; const float* bq  = (const float*)d_in[27];

  const int NJ  = in_sizes[0] / SBERT;
  const int NS  = in_sizes[1] / SBERT;
  const int EJS = in_sizes[2], ESJ = in_sizes[4], ESS = in_sizes[6];

  float* ws = (float*)d_ws;
  size_t off = 0;
  auto alloc = [&](size_t n) { float* p = ws + off; off += n; return p; };
  float* XJ     = alloc((size_t)NJ * HID);
  float* XS     = alloc((size_t)NS * HID);
  float* HS_JS  = alloc((size_t)NJ * HH);
  float* HS_SJ  = alloc((size_t)NS * HH);
  float* HS_SS  = alloc((size_t)NS * HH);
  float* ACC_JS = alloc((size_t)NS * HH);
  float* ACC_SS = alloc((size_t)NS * HH);
  float* ACC_SJ = alloc((size_t)NJ * HH);
  float* AS_JS  = alloc((size_t)NJ * 2); float* AD_JS = alloc((size_t)NS * 2);
  float* AS_SJ  = alloc((size_t)NS * 2); float* AD_SJ = alloc((size_t)NJ * 2);
  float* AS_SS  = alloc((size_t)NS * 2); float* AD_SS = alloc((size_t)NS * 2);
  unsigned* M_JS = (unsigned*)alloc((size_t)NS * 2);
  unsigned* M_SJ = (unsigned*)alloc((size_t)NJ * 2);
  unsigned* M_SS = (unsigned*)alloc((size_t)NS * 2);
  float* DEN_JS = alloc((size_t)NS * 2);
  float* DEN_SJ = alloc((size_t)NJ * 2);
  float* DEN_SS = alloc((size_t)NS * 2);
  float* AV     = alloc(256);
  // Pre-packed bf16 weight fragments
  const size_t SZ_384x128 = (384 / 32) * (128 / 16) * 256;  // 24576 dwords
  const size_t SZ_128x256 = (128 / 32) * (256 / 16) * 256;  // 16384
  const size_t SZ_256x128 = (256 / 32) * (128 / 16) * 256;  // 16384
  const size_t SZ_128x128 = (128 / 32) * (128 / 16) * 256;  //  8192
  unsigned* WP0J = (unsigned*)alloc(SZ_384x128);
  unsigned* WP0S = (unsigned*)alloc(SZ_384x128);
  unsigned* WPS[6];
  for (int r = 0; r < 6; ++r) WPS[r] = (unsigned*)alloc(SZ_128x256);
  unsigned* WPI[4];
  for (int r = 0; r < 4; ++r) WPI[r] = (unsigned*)alloc(SZ_256x128);
  unsigned* WPJF = (unsigned*)alloc(SZ_128x128);

  float* outp   = (float*)d_out;
  float* scores = outp;
  float* jemb   = outp + NJ;
  float* qv     = outp + NJ + (size_t)NJ * 128;

  auto ggrid = [](int M) { return dim3((unsigned)((M + 63) / 64)); };
  auto tgrid = [](int n) { return dim3((unsigned)((n + 255) / 256)); };

  // Repack all GEMM weights once (f32 -> bf16 fragment layout)
  repack_w<<<tgrid((int)SZ_384x128), 256, 0, stream>>>(W0j, WP0J, SBERT, HID);
  repack_w<<<tgrid((int)SZ_384x128), 256, 0, stream>>>(W0s, WP0S, SBERT, HID);
  for (int r = 0; r < 6; ++r)
    repack_w<<<tgrid((int)SZ_128x256), 256, 0, stream>>>(gWs + (size_t)r * HID * HH, WPS[r], HID, HH);
  for (int r = 0; r < 4; ++r)
    repack_w<<<tgrid((int)SZ_256x128), 256, 0, stream>>>(iW + (size_t)r * HH * HID, WPI[r], HH, HID);
  repack_w<<<tgrid((int)SZ_128x128), 256, 0, stream>>>(Wjf, WPJF, HID, HID);

  // Initial per-type linear + LayerNorm + ReLU
  wmma_gemm<SBERT, HID, 2><<<ggrid(NJ), 128, 0, stream>>>(x_job,   WP0J, b0j, g0j, be0j, XJ, NJ);
  wmma_gemm<SBERT, HID, 2><<<ggrid(NS), 128, 0, stream>>>(x_skill, WP0S, b0s, g0s, be0s, XS, NS);

  for (int i = 0; i < 2; ++i) {
    const float* Wd[3]; const float* as_[3]; const float* ad_[3]; const float* bb[3];
    for (int r = 0; r < 3; ++r) {
      Wd[r]  = gWd + (size_t)(i * 3 + r) * HID * HH;
      as_[r] = gas + (size_t)(i * 3 + r) * 2 * HID;
      ad_[r] = gad + (size_t)(i * 3 + r) * 2 * HID;
      bb[r]  = gb  + (size_t)(i * 3 + r) * HH;
    }
    // Source projections (no bias in GAT linears)
    wmma_gemm<HID, HH, 0><<<ggrid(NJ), 128, 0, stream>>>(XJ, WPS[i * 3 + 0], nullptr, nullptr, nullptr, HS_JS, NJ);
    wmma_gemm<HID, HH, 0><<<ggrid(NS), 128, 0, stream>>>(XS, WPS[i * 3 + 1], nullptr, nullptr, nullptr, HS_SJ, NS);
    wmma_gemm<HID, HH, 0><<<ggrid(NS), 128, 0, stream>>>(XS, WPS[i * 3 + 2], nullptr, nullptr, nullptr, HS_SS, NS);
    // alpha_src from projections
    alpha_k<<<tgrid(NJ * 2), 256, 0, stream>>>(HS_JS, HH, HID, as_[0], AS_JS, NJ * 2);
    alpha_k<<<tgrid(NS * 2), 256, 0, stream>>>(HS_SJ, HH, HID, as_[1], AS_SJ, NS * 2);
    alpha_k<<<tgrid(NS * 2), 256, 0, stream>>>(HS_SS, HH, HID, as_[2], AS_SS, NS * 2);
    // alpha_dst via folded Wd@a_d (avoids materializing hd)
    fold_vec<<<1, 256, 0, stream>>>(Wd[0], ad_[0], AV);
    alpha_k<<<tgrid(NS * 2), 256, 0, stream>>>(XS, HID, 0, AV, AD_JS, NS * 2);
    fold_vec<<<1, 256, 0, stream>>>(Wd[1], ad_[1], AV);
    alpha_k<<<tgrid(NJ * 2), 256, 0, stream>>>(XJ, HID, 0, AV, AD_SJ, NJ * 2);
    fold_vec<<<1, 256, 0, stream>>>(Wd[2], ad_[2], AV);
    alpha_k<<<tgrid(NS * 2), 256, 0, stream>>>(XS, HID, 0, AV, AD_SS, NS * 2);
    // Init softmax state + accumulators
    fill_u32<<<tgrid(NS * 2), 256, 0, stream>>>(M_JS, 0u, NS * 2);
    fill_u32<<<tgrid(NJ * 2), 256, 0, stream>>>(M_SJ, 0u, NJ * 2);
    fill_u32<<<tgrid(NS * 2), 256, 0, stream>>>(M_SS, 0u, NS * 2);
    fill_u32<<<tgrid(NS * 2), 256, 0, stream>>>((unsigned*)DEN_JS, 0u, NS * 2);
    fill_u32<<<tgrid(NJ * 2), 256, 0, stream>>>((unsigned*)DEN_SJ, 0u, NJ * 2);
    fill_u32<<<tgrid(NS * 2), 256, 0, stream>>>((unsigned*)DEN_SS, 0u, NS * 2);
    fill_u32<<<tgrid(NS * HH), 256, 0, stream>>>((unsigned*)ACC_JS, 0u, NS * HH);
    fill_u32<<<tgrid(NS * HH), 256, 0, stream>>>((unsigned*)ACC_SS, 0u, NS * HH);
    fill_u32<<<tgrid(NJ * HH), 256, 0, stream>>>((unsigned*)ACC_SJ, 0u, NJ * HH);
    // Edge pass 1: scatter max (ss relation appends NS self-loops in-kernel)
    edge_max<<<tgrid(EJS * 2), 256, 0, stream>>>(js_src, js_dst, AS_JS, AD_JS, M_JS, EJS, EJS);
    edge_max<<<tgrid(ESJ * 2), 256, 0, stream>>>(sj_src, sj_dst, AS_SJ, AD_SJ, M_SJ, ESJ, ESJ);
    edge_max<<<tgrid((ESS + NS) * 2), 256, 0, stream>>>(ss_src, ss_dst, AS_SS, AD_SS, M_SS, ESS, ESS + NS);
    // Edge pass 2: exp / denominator / weighted scatter (one wave per edge)
    edge_accum<<<dim3((unsigned)((EJS + 7) / 8)), 256, 0, stream>>>(js_src, js_dst, AS_JS, AD_JS, M_JS, DEN_JS, HS_JS, ACC_JS, EJS, EJS);
    edge_accum<<<dim3((unsigned)((ESJ + 7) / 8)), 256, 0, stream>>>(sj_src, sj_dst, AS_SJ, AD_SJ, M_SJ, DEN_SJ, HS_SJ, ACC_SJ, ESJ, ESJ);
    edge_accum<<<dim3((unsigned)((ESS + NS + 7) / 8)), 256, 0, stream>>>(ss_src, ss_dst, AS_SS, AD_SS, M_SS, DEN_SS, HS_SS, ACC_SS, ESS, ESS + NS);
    // Normalize + bias; HeteroConv sum for skill (js + ss). In-place into ACC buffers.
    node_final<<<tgrid(NS * HH), 256, 0, stream>>>(ACC_JS, DEN_JS, bb[0], ACC_SS, DEN_SS, bb[2], ACC_JS, NS);
    node_final<<<tgrid(NJ * HH), 256, 0, stream>>>(ACC_SJ, DEN_SJ, bb[1], nullptr, nullptr, nullptr, ACC_SJ, NJ);
    // Inter linear + ReLU -> next layer features
    wmma_gemm<HH, HID, 1><<<ggrid(NJ), 128, 0, stream>>>(ACC_SJ, WPI[i * 2 + 0], ib + (size_t)(i * 2 + 0) * HID, nullptr, nullptr, XJ, NJ);
    wmma_gemm<HH, HID, 1><<<ggrid(NS), 128, 0, stream>>>(ACC_JS, WPI[i * 2 + 1], ib + (size_t)(i * 2 + 1) * HID, nullptr, nullptr, XS, NS);
  }

  // Final head: job_emb, q, scores
  wmma_gemm<HID, HID, 0><<<ggrid(NJ), 128, 0, stream>>>(XJ, WPJF, bjf, nullptr, nullptr, jemb, NJ);
  qproj<<<1, 128, 0, stream>>>(query, Wq, bq, qv);
  scores_k<<<tgrid(NJ), 256, 0, stream>>>(jemb, qv, scores, NJ);
}
